// TinyTransformerLayer_27771258536881
// MI455X (gfx1250) — matrix operand order repaired and checked by reading, not compiled, not run on hardware
//
#include <hip/hip_runtime.h>
#include <math.h>

// ---------------------------------------------------------------------------
// Types for CDNA5 WMMA (wave32): v_wmma_f32_16x16x32_bf16
// ---------------------------------------------------------------------------
typedef __attribute__((ext_vector_type(16))) __bf16          v16bf;
typedef __attribute__((ext_vector_type(16))) unsigned short  v16u;
typedef __attribute__((ext_vector_type(8)))  unsigned short  v8u;
typedef __attribute__((ext_vector_type(8)))  float           v8f;

union BCast16 { v16u u; v16bf b; };
union AJoin   { struct { v8u lo, hi; } s; v16u v; };

__device__ __forceinline__ unsigned short f2bf(float f) {
  unsigned int u = __float_as_uint(f);
  unsigned int r = u + 0x7fffu + ((u >> 16) & 1u);   // round-to-nearest-even
  return (unsigned short)(r >> 16);
}

__device__ __forceinline__ v8f wmma_bf16(v16bf a, v16bf b, v8f c) {
  return __builtin_amdgcn_wmma_f32_16x16x32_bf16(false, a, false, b,
                                                 (short)0, c, false, false);
}

// ---------------------------------------------------------------------------
// f32 -> bf16 conversion
// ---------------------------------------------------------------------------
__global__ void cvt_f32_bf16(const float* __restrict__ in,
                             unsigned short* __restrict__ out, int n) {
  int i = blockIdx.x * blockDim.x + threadIdx.x;
  int stride = gridDim.x * blockDim.x;
  for (; i < n; i += stride) out[i] = f2bf(in[i]);
}

// ---------------------------------------------------------------------------
// C[M,N] = Abf[M,K] @ Bbf[N,K]^T + bias     (bf16 inputs, f32 accumulate)
// 128 threads = 4 waves; block tile 128x64; wave tile 32x64:
//   2 A-fragments x 4 B-fragments = 8 WMMA per 32-wide k-step,
//   every B-fragment reused twice from registers.
// ---------------------------------------------------------------------------
template <bool BF16OUT>
__global__ __launch_bounds__(128)
void gemm_bf16(const unsigned short* __restrict__ A,
               const unsigned short* __restrict__ B,
               const float* __restrict__ bias,
               unsigned short* __restrict__ out_bf,
               float* __restrict__ out_f,
               int M, int N, int K) {
  const int tid  = threadIdx.x;
  const int wave = tid >> 5;
  const int lane = tid & 31;
  const int llow = lane & 15;
  const int hs   = lane >> 4;            // which half of the wave
  const int mbase = blockIdx.y * 128 + wave * 32;
  const int nbase = blockIdx.x * 64;

  v8f acc[2][4];
  #pragma unroll
  for (int r = 0; r < 2; ++r)
    #pragma unroll
    for (int j = 0; j < 4; ++j) acc[r][j] = (v8f)0.0f;

  const unsigned short* arow0 = A + (size_t)(mbase + llow) * K;
  const unsigned short* arow1 = A + (size_t)(mbase + 16 + llow) * K;

  for (int kk = 0; kk < K; kk += 32) {
    // A fragments (16x32 bf16): lane L: K = hs*16+0..15 (contiguous)
    AJoin aj0, aj1;
    aj0.s.lo = *(const v8u*)(arow0 + kk + hs * 16);
    aj0.s.hi = *(const v8u*)(arow0 + kk + hs * 16 + 8);
    aj1.s.lo = *(const v8u*)(arow1 + kk + hs * 16);
    aj1.s.hi = *(const v8u*)(arow1 + kk + hs * 16 + 8);
    BCast16 a0, a1; a0.u = aj0.v; a1.u = aj1.v;
    #pragma unroll
    for (int j = 0; j < 4; ++j) {
      // B fragment (32x16): lane n holds B[k,n] = Brow[n][k0 + hs*16 .. +16]
      const unsigned short* brow =
          B + (size_t)(nbase + j * 16 + llow) * K + kk + hs * 16;
      BCast16 bc; bc.u = *(const v16u*)brow;
      acc[0][j] = wmma_bf16(a0.b, bc.b, acc[0][j]);
      acc[1][j] = wmma_bf16(a1.b, bc.b, acc[1][j]);
    }
  }

  // Epilogue: C layout -> element (m,n): vgpr v = m%8, lane = n%16 + 16*(m/8)
  #pragma unroll
  for (int r = 0; r < 2; ++r)
    #pragma unroll
    for (int j = 0; j < 4; ++j) {
      const int col = nbase + j * 16 + llow;
      const float bv = bias[col];
      #pragma unroll
      for (int v = 0; v < 8; ++v) {
        const int row = mbase + r * 16 + v + 8 * hs;
        const float val = acc[r][j][v] + bv;
        if (BF16OUT) out_bf[(size_t)row * N + col] = f2bf(val);
        else         out_f [(size_t)row * N + col] = val;
      }
    }
}

// ---------------------------------------------------------------------------
// Flash attention: qkv bf16 [B*S, 3072] (per head h: q=h*192, k=+64, v=+128)
// -> outbf [B*S, 1024] (col = h*64 + d).  grid = (S/64, NH, B), 128 threads.
// Each wave: 16 q-rows, online softmax over 32 KV tiles of 64.
// V is staged TRANSPOSED (d-major) in LDS so P@V B-fragments are contiguous
// ds_load_b128 pairs instead of 16-element scalar gathers.
// ---------------------------------------------------------------------------
#define SEQ_   2048
#define NH_    16
#define QLD_   3072
#define PSTR   72      // padded P-tile row stride (elements)
#define VTSTR  72      // padded transposed-V row stride (elements)

__global__ __launch_bounds__(128)
void flash_attn(const unsigned short* __restrict__ qkv,
                unsigned short* __restrict__ outbf) {
  const int qtile = blockIdx.x;
  const int h     = blockIdx.y;
  const int b     = blockIdx.z;
  const int tid   = threadIdx.x;
  const int wave  = tid >> 5;
  const int lane  = tid & 31;
  const int llow  = lane & 15;
  const int hs    = lane >> 4;

  const size_t rowbase = (size_t)b * SEQ_;
  const int qbase = qtile * 64 + wave * 16;

  __shared__ unsigned short Vt[64 * VTSTR];      // V tile, transposed: [d][kv]
  __shared__ unsigned short Pt[4][16 * PSTR];    // per-wave P staging

  // Preload Q fragments (A-layout), constant over the KV loop. K dim = 64.
  v16bf qf[2];
  #pragma unroll
  for (int kc = 0; kc < 2; ++kc) {
    const unsigned short* qrow =
        qkv + (rowbase + qbase + llow) * QLD_ + h * 192 + kc * 32;
    AJoin aj;
    aj.s.lo = *(const v8u*)(qrow + hs * 16);
    aj.s.hi = *(const v8u*)(qrow + hs * 16 + 8);
    BCast16 c; c.u = aj.v; qf[kc] = c.b;
  }

  v8f o[4];
  #pragma unroll
  for (int j = 0; j < 4; ++j) o[j] = (v8f)0.0f;
  float mi[8], li[8];
  #pragma unroll
  for (int v = 0; v < 8; ++v) { mi[v] = -__builtin_inff(); li[v] = 0.0f; }

  for (int t = 0; t < SEQ_ / 64; ++t) {
    const int kvbase = t * 64;
    __syncthreads();                       // Vt free to overwrite
    // Cooperative V tile load, transposed into LDS: Vt[d][kv]
    {
      const int r  = tid >> 1;             // kv row 0..63
      const int c0 = (tid & 1) * 32;       // d col base
      const unsigned short* src =
          qkv + (rowbase + kvbase + r) * QLD_ + h * 192 + 128 + c0;
      v16u b0 = *(const v16u*)(src);
      v16u b1 = *(const v16u*)(src + 16);
      #pragma unroll
      for (int i = 0; i < 16; ++i) Vt[(c0 + i) * VTSTR + r]      = b0[i];
      #pragma unroll
      for (int i = 0; i < 16; ++i) Vt[(c0 + 16 + i) * VTSTR + r] = b1[i];
    }

    // S = Q @ K^T  (16 q-rows x 64 kv-cols per wave)
    v8f s[4];
    #pragma unroll
    for (int j = 0; j < 4; ++j) s[j] = (v8f)0.0f;
    #pragma unroll
    for (int j = 0; j < 4; ++j) {
      #pragma unroll
      for (int kc = 0; kc < 2; ++kc) {
        const unsigned short* krow =
            qkv + (rowbase + kvbase + j * 16 + llow) * QLD_ +
            h * 192 + 64 + kc * 32 + hs * 16;
        BCast16 bc; bc.u = *(const v16u*)krow;
        s[j] = wmma_bf16(qf[kc], bc.b, s[j]);
      }
    }
    __syncthreads();                       // Vt ready for all waves

    // scale
    #pragma unroll
    for (int j = 0; j < 4; ++j)
      #pragma unroll
      for (int v = 0; v < 8; ++v) s[j][v] *= 0.125f;

    // Row max across 64 kv columns (4 accs x 16 lanes within half-wave)
    float mt[8];
    #pragma unroll
    for (int v = 0; v < 8; ++v)
      mt[v] = fmaxf(fmaxf(s[0][v], s[1][v]), fmaxf(s[2][v], s[3][v]));
    #pragma unroll
    for (int off = 8; off >= 1; off >>= 1)
      #pragma unroll
      for (int v = 0; v < 8; ++v)
        mt[v] = fmaxf(mt[v], __shfl_xor(mt[v], off, 16));

    float alpha[8];
    #pragma unroll
    for (int v = 0; v < 8; ++v) {
      const float mn = fmaxf(mi[v], mt[v]);
      alpha[v] = __expf(mi[v] - mn);
      mi[v] = mn;
    }
    #pragma unroll
    for (int j = 0; j < 4; ++j)
      #pragma unroll
      for (int v = 0; v < 8; ++v) s[j][v] = __expf(s[j][v] - mi[v]);

    float rs[8];
    #pragma unroll
    for (int v = 0; v < 8; ++v) rs[v] = s[0][v] + s[1][v] + s[2][v] + s[3][v];
    #pragma unroll
    for (int off = 8; off >= 1; off >>= 1)
      #pragma unroll
      for (int v = 0; v < 8; ++v) rs[v] += __shfl_xor(rs[v], off, 16);
    #pragma unroll
    for (int v = 0; v < 8; ++v) li[v] = li[v] * alpha[v] + rs[v];
    #pragma unroll
    for (int j = 0; j < 4; ++j)
      #pragma unroll
      for (int v = 0; v < 8; ++v) o[j][v] *= alpha[v];

    // Stage P (bf16) to LDS in row-major for A-fragment re-read
    #pragma unroll
    for (int j = 0; j < 4; ++j)
      #pragma unroll
      for (int v = 0; v < 8; ++v)
        Pt[wave][(v + 8 * hs) * PSTR + j * 16 + llow] = f2bf(s[j][v]);
    __syncthreads();

    // O += P @ V  (K = 64 -> two 16x16x32 steps per 16-col chunk of V)
    #pragma unroll
    for (int kc = 0; kc < 2; ++kc) {
      AJoin aj;
      aj.s.lo = *(const v8u*)&Pt[wave][llow * PSTR + kc * 32 + hs * 16];
      aj.s.hi = *(const v8u*)&Pt[wave][llow * PSTR + kc * 32 + hs * 16 + 8];
      BCast16 pa; pa.u = aj.v;
      #pragma unroll
      for (int j = 0; j < 4; ++j) {
        // B-frag from transposed V: lane n holds Vt[j*16+n][kc*32+hs*16 ..+16]
        const unsigned short* vr = &Vt[(j * 16 + llow) * VTSTR + kc * 32 + hs * 16];
        AJoin vj;
        vj.s.lo = *(const v8u*)(vr);
        vj.s.hi = *(const v8u*)(vr + 8);
        BCast16 vb; vb.u = vj.v;
        o[j] = wmma_bf16(pa.b, vb.b, o[j]);
      }
    }
  }

  // Normalize and store bf16 attention output: col = h*64 + j*16 + llow
  float inv[8];
  #pragma unroll
  for (int v = 0; v < 8; ++v) inv[v] = 1.0f / li[v];
  #pragma unroll
  for (int j = 0; j < 4; ++j)
    #pragma unroll
    for (int v = 0; v < 8; ++v) {
      const size_t row = rowbase + qbase + v + 8 * hs;
      outbf[row * 1024 + h * 64 + j * 16 + llow] = f2bf(o[j][v] * inv[v]);
    }
}

// ---------------------------------------------------------------------------
// Host launcher
// ---------------------------------------------------------------------------
extern "C" void kernel_launch(void* const* d_in, const int* in_sizes, int n_in,
                              void* d_out, int out_size, void* d_ws, size_t ws_size,
                              hipStream_t stream) {
  (void)in_sizes; (void)n_in; (void)out_size; (void)ws_size;
  const float* x     = (const float*)d_in[0];   // (2,2048,1024)
  const float* W_qkv = (const float*)d_in[1];   // (3072,1024)
  const float* b_qkv = (const float*)d_in[2];   // (3072,)
  const float* W_out = (const float*)d_in[3];   // (1024,1024)
  const float* b_out = (const float*)d_in[4];   // (1024,)
  float* out = (float*)d_out;

  const int M = 4096, D = 1024, N3 = 3072;

  char* ws = (char*)d_ws;
  unsigned short* Xbf   = (unsigned short*)(ws);
  unsigned short* Wqkvb = (unsigned short*)(ws + (size_t)8  * 1024 * 1024);
  unsigned short* Woutb = (unsigned short*)(ws + (size_t)14 * 1024 * 1024);
  unsigned short* QKVb  = (unsigned short*)(ws + (size_t)16 * 1024 * 1024);
  unsigned short* Attnb = (unsigned short*)(ws + (size_t)40 * 1024 * 1024);

  cvt_f32_bf16<<<1024, 256, 0, stream>>>(x,     Xbf,   M * D);
  cvt_f32_bf16<<<1024, 256, 0, stream>>>(W_qkv, Wqkvb, N3 * D);
  cvt_f32_bf16<<<1024, 256, 0, stream>>>(W_out, Woutb, D * D);

  // qkv = x @ W_qkv^T + b_qkv  -> bf16
  gemm_bf16<true><<<dim3(N3 / 64, M / 128), 128, 0, stream>>>(
      Xbf, Wqkvb, b_qkv, QKVb, nullptr, M, N3, D);

  // flash attention per (qtile, head, batch)
  flash_attn<<<dim3(SEQ_ / 64, NH_, 2), 128, 0, stream>>>(QKVb, Attnb);

  // out = attn @ W_out^T + b_out -> f32
  gemm_bf16<false><<<dim3(D / 64, M / 128), 128, 0, stream>>>(
      Attnb, Woutb, b_out, nullptr, out, M, D, D);
}
